// MimiAttention_18983755448884
// MI455X (gfx1250) — compile-verified, matrix operand order checked by reading, not run on hardware
//
#include <hip/hip_runtime.h>
#include <hip/hip_bf16.h>

#define T    4096
#define HID  1024
#define NH   16
#define HD   64
#define WINDOW 512

typedef __bf16 v16bf __attribute__((ext_vector_type(16)));
typedef __bf16 v8bf  __attribute__((ext_vector_type(8)));
typedef float  v8f   __attribute__((ext_vector_type(8)));

// Load a 16x32 bf16 WMMA fragment (A layout; B layout is the mirrored case and
// uses the same code with "row" = N column). `p` points at
// base + row*stride + koff, where koff = (lane>=16) ? 8 : 0.
// Halves 0..7  <- K = koff+0..7      (16B contiguous)
// Halves 8..15 <- K = 16+koff+0..7   (16B contiguous)
__device__ __forceinline__ v16bf load_frag16x32(const __bf16* p) {
    v8bf lo = *(const v8bf*)(p);
    v8bf hi = *(const v8bf*)(p + 16);
    v16bf r;
#pragma unroll
    for (int i = 0; i < 8; ++i) { r[i] = lo[i]; r[i + 8] = hi[i]; }
    return r;
}

__device__ __forceinline__ v8f wmma_bf16(v16bf a, v16bf b, v8f c) {
    return __builtin_amdgcn_wmma_f32_16x16x32_bf16(
        /*neg_a=*/false, a, /*neg_b=*/false, b,
        /*c_mod=*/(short)0, c, /*reuse_a=*/false, /*reuse_b=*/false);
}

// ---------------------------------------------------------------------------
// f32 -> bf16 cast (one-time; working set then lives bf16 in L2)
// ---------------------------------------------------------------------------
__global__ void f32_to_bf16(const float* __restrict__ in,
                            __bf16* __restrict__ out, int n) {
    int i = blockIdx.x * blockDim.x + threadIdx.x;
    if (i < n) out[i] = (__bf16)in[i];
}

// ---------------------------------------------------------------------------
// C(f32, MxN) = A(bf16, MxK row-major) @ W(bf16, NxK row-major)^T
// Block: 128 threads = 4 waves; block tile 128(M)x64(N).
// Each wave owns a 32(M)x64(N) tile: 8 f32 accumulators; per k-step (K=32):
// 4 A loads + 8 B loads feed 8 WMMAs (B fragment shared across 2 A frags).
// global_prefetch of the k+128 fragments keeps the WGP cache primed.
// ---------------------------------------------------------------------------
__global__ void gemm_xwT(const __bf16* __restrict__ A,
                         const __bf16* __restrict__ W,
                         float* __restrict__ C,
                         int M, int N, int K) {
    const int lane  = threadIdx.x & 31;
    const int wave  = threadIdx.x >> 5;
    const int rr    = lane & 15;
    const int koff  = (lane >> 4) * 8;
    const int rAdd  = (lane >= 16) ? 8 : 0;
    const int m0    = blockIdx.y * 128 + wave * 32;
    const int n0    = blockIdx.x * 64;

    v8f zero = {0.f,0.f,0.f,0.f,0.f,0.f,0.f,0.f};
    v8f acc[2][4];
#pragma unroll
    for (int mt = 0; mt < 2; ++mt)
#pragma unroll
        for (int nt = 0; nt < 4; ++nt) acc[mt][nt] = zero;

    const __bf16* aRow0 = A + (size_t)(m0 + rr) * K + koff;
    const __bf16* aRow1 = A + (size_t)(m0 + 16 + rr) * K + koff;
    const __bf16* bRow[4];
#pragma unroll
    for (int nt = 0; nt < 4; ++nt)
        bRow[nt] = W + (size_t)(n0 + nt * 16 + rr) * K + koff;

    for (int kb = 0; kb < K; kb += 32) {
        if (kb + 128 < K) {               // wave-uniform: EXEC stays all-ones
            __builtin_prefetch(aRow0 + kb + 128, 0, 0);
            __builtin_prefetch(aRow1 + kb + 128, 0, 0);
            __builtin_prefetch(bRow[0] + kb + 128, 0, 0);
            __builtin_prefetch(bRow[2] + kb + 128, 0, 0);
        }
        v16bf a0 = load_frag16x32(aRow0 + kb);
        v16bf a1 = load_frag16x32(aRow1 + kb);
#pragma unroll
        for (int nt = 0; nt < 4; ++nt) {
            v16bf bf = load_frag16x32(bRow[nt] + kb);
            acc[0][nt] = wmma_bf16(a0, bf, acc[0][nt]);
            acc[1][nt] = wmma_bf16(a1, bf, acc[1][nt]);
        }
    }
#pragma unroll
    for (int mt = 0; mt < 2; ++mt) {
#pragma unroll
        for (int nt = 0; nt < 4; ++nt) {
            const int col = n0 + nt * 16 + rr;
#pragma unroll
            for (int r = 0; r < 8; ++r)
                C[(size_t)(m0 + mt * 16 + r + rAdd) * N + col] = acc[mt][nt][r];
        }
    }
}

// ---------------------------------------------------------------------------
// RoPE on Q and K (f32 in) -> bf16 out. One thread per (t, h, i<32) pair.
// ---------------------------------------------------------------------------
__global__ void rope_convert(const float* __restrict__ Qf,
                             const float* __restrict__ Kf,
                             __bf16* __restrict__ Qb,
                             __bf16* __restrict__ Kb) {
    int idx = blockIdx.x * blockDim.x + threadIdx.x;
    if (idx >= T * NH * (HD / 2)) return;
    const int i = idx & 31;
    const int h = (idx >> 5) & (NH - 1);
    const int t = idx >> 9;

    const float inv = powf(10000.0f, -(float)(2 * i) / (float)HD);
    float s, c;
    sincosf((float)t * inv, &s, &c);

    const int base = t * HID + h * HD;
    const float q1 = Qf[base + i], q2 = Qf[base + i + 32];
    Qb[base + i]      = (__bf16)(q1 * c - q2 * s);
    Qb[base + i + 32] = (__bf16)(q2 * c + q1 * s);
    const float k1 = Kf[base + i], k2 = Kf[base + i + 32];
    Kb[base + i]      = (__bf16)(k1 * c - k2 * s);
    Kb[base + i + 32] = (__bf16)(k2 * c + k1 * s);
}

// ---------------------------------------------------------------------------
// V (f32, [t][h*64+d]) -> Vt (bf16, [h*64+d][t]); coalesced stores.
// Transposed V makes the P@V B-fragment two contiguous b128 loads.
// ---------------------------------------------------------------------------
__global__ void v_transpose(const float* __restrict__ Vf,
                            __bf16* __restrict__ Vt) {
    int idx = blockIdx.x * blockDim.x + threadIdx.x;
    if (idx >= T * HID) return;
    const int t = idx & (T - 1);
    const int d = idx >> 12;              // T == 4096
    Vt[idx] = (__bf16)Vf[(size_t)t * HID + d];
}

// ---------------------------------------------------------------------------
// Flash attention, 512-key causal window. One wave per (16-query tile, head).
// S = Q Kt via 2 WMMAs per 16-key tile (HD=64 -> two K=32 steps);
// online softmax in registers (shfl_xor row reductions over half-waves);
// P staged in 1KB LDS to relayout C->A; O += P V via 4 WMMAs per 32 keys.
// ---------------------------------------------------------------------------
__global__ void attn_local(const __bf16* __restrict__ Qb,
                           const __bf16* __restrict__ Kb,
                           const __bf16* __restrict__ Vt,
                           __bf16* __restrict__ ctx) {
    __shared__ __align__(16) __bf16 Pst[16][32];

    const int lane = threadIdx.x & 31;
    const int rr   = lane & 15;
    const int koff = (lane >> 4) * 8;
    const int rAdd = (lane >= 16) ? 8 : 0;
    const int q0   = blockIdx.x * 16;
    const int h    = blockIdx.y;

    // Q fragments (d-steps 0..31 and 32..63), loaded once.
    const __bf16* qrow = Qb + (size_t)(q0 + rr) * HID + h * HD + koff;
    const v16bf aq0 = load_frag16x32(qrow);
    const v16bf aq1 = load_frag16x32(qrow + 32);

    v8f zero = {0.f,0.f,0.f,0.f,0.f,0.f,0.f,0.f};
    v8f o[4];
#pragma unroll
    for (int nt = 0; nt < 4; ++nt) o[nt] = zero;
    float m[8], l[8];
#pragma unroll
    for (int r = 0; r < 8; ++r) { m[r] = -3.0e38f; l[r] = 0.f; }

    int kstart = q0 - WINDOW; if (kstart < 0) kstart = 0;
    kstart &= ~31;
    const int kend = q0 + 15;
    const float scale = 0.125f;           // 1/sqrt(64)

    for (int kb = kstart; kb <= kend; kb += 32) {
        float sv[2][8];
#pragma unroll
        for (int tile = 0; tile < 2; ++tile) {
            const __bf16* krow =
                Kb + (size_t)(kb + tile * 16 + rr) * HID + h * HD + koff;
            v16bf bk0 = load_frag16x32(krow);
            v16bf bk1 = load_frag16x32(krow + 32);
            v8f sacc = wmma_bf16(aq0, bk0, zero);
            sacc     = wmma_bf16(aq1, bk1, sacc);
            const int kt = kb + tile * 16 + rr;   // this lane's key column
#pragma unroll
            for (int r = 0; r < 8; ++r) {
                const int qt = q0 + r + rAdd;
                float s = sacc[r] * scale;
                if (kt > qt || kt < qt - WINDOW) s = -1.0e30f;
                sv[tile][r] = s;
            }
        }
        // Online softmax update (row r lives in one 16-lane half-wave).
#pragma unroll
        for (int r = 0; r < 8; ++r) {
            float rm = fmaxf(sv[0][r], sv[1][r]);
#pragma unroll
            for (int mk = 8; mk >= 1; mk >>= 1)
                rm = fmaxf(rm, __shfl_xor(rm, mk, 32));
            const float mnew = fmaxf(m[r], rm);
            const float corr = __expf(m[r] - mnew);
            m[r] = mnew;
            l[r] *= corr;
#pragma unroll
            for (int nt = 0; nt < 4; ++nt) o[nt][r] *= corr;
            const float p0 = (sv[0][r] <= -1.0e29f) ? 0.f : __expf(sv[0][r] - mnew);
            const float p1 = (sv[1][r] <= -1.0e29f) ? 0.f : __expf(sv[1][r] - mnew);
            float ps = p0 + p1;
#pragma unroll
            for (int mk = 8; mk >= 1; mk >>= 1)
                ps += __shfl_xor(ps, mk, 32);
            l[r] += ps;
            Pst[r + rAdd][rr]      = (__bf16)p0;
            Pst[r + rAdd][16 + rr] = (__bf16)p1;
        }
        __syncthreads();
        // Relayout P (C-layout -> A-layout) through LDS, then O += P @ V.
        v16bf pf = load_frag16x32(&Pst[rr][koff]);
#pragma unroll
        for (int nt = 0; nt < 4; ++nt) {
            const __bf16* vrow =
                Vt + (size_t)(h * HD + nt * 16 + rr) * T + kb + koff;
            v16bf bv = load_frag16x32(vrow);
            o[nt] = wmma_bf16(pf, bv, o[nt]);
        }
        __syncthreads();
    }
    // Normalize and store ctx as bf16 [t][h*64+d] for the output projection.
#pragma unroll
    for (int nt = 0; nt < 4; ++nt) {
#pragma unroll
        for (int r = 0; r < 8; ++r) {
            const float val = o[nt][r] / l[r];
            ctx[(size_t)(q0 + r + rAdd) * HID + h * HD + nt * 16 + rr] =
                (__bf16)val;
        }
    }
}

// ---------------------------------------------------------------------------
extern "C" void kernel_launch(void* const* d_in, const int* in_sizes, int n_in,
                              void* d_out, int out_size, void* d_ws, size_t ws_size,
                              hipStream_t stream) {
    const float* x  = (const float*)d_in[0];
    const float* wq = (const float*)d_in[1];
    const float* wk = (const float*)d_in[2];
    const float* wv = (const float*)d_in[3];
    const float* wo = (const float*)d_in[4];
    float* out = (float*)d_out;

    char* w = (char*)d_ws;
    auto alloc = [&](size_t bytes) -> void* {
        void* p = (void*)w;
        w += (bytes + 255) & ~(size_t)255;
        return p;
    };
    __bf16* xb   = (__bf16*)alloc((size_t)T * HID * 2);
    __bf16* wqb  = (__bf16*)alloc((size_t)HID * HID * 2);
    __bf16* wkb  = (__bf16*)alloc((size_t)HID * HID * 2);
    __bf16* wvb  = (__bf16*)alloc((size_t)HID * HID * 2);
    __bf16* wob  = (__bf16*)alloc((size_t)HID * HID * 2);
    float*  Qf   = (float*) alloc((size_t)T * HID * 4);
    float*  Kf   = (float*) alloc((size_t)T * HID * 4);
    float*  Vf   = (float*) alloc((size_t)T * HID * 4);
    __bf16* Qb   = (__bf16*)alloc((size_t)T * HID * 2);
    __bf16* Kb2  = (__bf16*)alloc((size_t)T * HID * 2);
    __bf16* Vt   = (__bf16*)alloc((size_t)T * HID * 2);
    __bf16* ctxb = (__bf16*)alloc((size_t)T * HID * 2);

    const dim3 cb(256);
    f32_to_bf16<<<(T * HID + 255) / 256, cb, 0, stream>>>(x,  xb,  T * HID);
    f32_to_bf16<<<(HID * HID + 255) / 256, cb, 0, stream>>>(wq, wqb, HID * HID);
    f32_to_bf16<<<(HID * HID + 255) / 256, cb, 0, stream>>>(wk, wkb, HID * HID);
    f32_to_bf16<<<(HID * HID + 255) / 256, cb, 0, stream>>>(wv, wvb, HID * HID);
    f32_to_bf16<<<(HID * HID + 255) / 256, cb, 0, stream>>>(wo, wob, HID * HID);

    const dim3 gb(HID / 64, T / 128), bb(128);
    gemm_xwT<<<gb, bb, 0, stream>>>(xb, wqb, Qf, T, HID, HID);
    gemm_xwT<<<gb, bb, 0, stream>>>(xb, wkb, Kf, T, HID, HID);
    gemm_xwT<<<gb, bb, 0, stream>>>(xb, wvb, Vf, T, HID, HID);

    rope_convert<<<(T * NH * (HD / 2) + 255) / 256, cb, 0, stream>>>(Qf, Kf, Qb, Kb2);
    v_transpose<<<(T * HID + 255) / 256, cb, 0, stream>>>(Vf, Vt);

    attn_local<<<dim3(T / 16, NH), dim3(32), 0, stream>>>(Qb, Kb2, Vt, ctxb);

    gemm_xwT<<<gb, bb, 0, stream>>>(ctxb, wob, out, T, HID, HID);
}
